// EquiLinear_18468359372804
// MI455X (gfx1250) — compile-verified, matrix operand order checked by reading, not run on hardware
//
#include <hip/hip_runtime.h>
#include <hip/hip_bf16.h>
#include <stdint.h>

// ---------------------------------------------------------------------------
// Problem constants
// ---------------------------------------------------------------------------
#define B_BATCH   128
#define N_ATOMS   512
#define N_CGS     16
#define NPAIR     256                 // N_CGS * N_CGS
#define KTOT      65792               // NPAIR + NPAIR*NPAIR
#define NCOLS     384                 // B_BATCH * 3
#define KITER_TOT 2056                // KTOT / 32  (32 = bf16 WMMA K)
#define KSPLIT    16
#define KITER_CH  129                 // ceil(2056/16); last chunk gets 121

typedef __attribute__((ext_vector_type(16))) __bf16 v16bf;
typedef __attribute__((ext_vector_type(8)))  float  v8f;

union Frag { uint4 u[2]; v16bf v; };

__device__ __forceinline__ unsigned short f2bf(float f) {
    // round-to-nearest-even fp32 -> bf16
    unsigned int u = __float_as_uint(f);
    unsigned int r = u + 0x7FFFu + ((u >> 16) & 1u);
    return (unsigned short)(r >> 16);
}

// ---------------------------------------------------------------------------
// 1) W (f32) -> bf16
// ---------------------------------------------------------------------------
__global__ void convert_w_kernel(const float* __restrict__ W,
                                 unsigned short* __restrict__ Wbf, int n) {
    int i = blockIdx.x * blockDim.x + threadIdx.x;
    if (i < n) Wbf[i] = f2bf(W[i]);
}

// ---------------------------------------------------------------------------
// 2) Build BbigT[n][k] (bf16, K-contiguous) from cg_xyz.
//    j <  256 : Bmat[p]      = cg[p%16] - cg[p/16]
//    j >= 256 : cross(Bmat[q], Bmat[p]) for j-256 = p*256+q
//    Row n = b*3+e.
// ---------------------------------------------------------------------------
__global__ void build_bbigT_kernel(const float* __restrict__ cg_xyz,
                                   unsigned short* __restrict__ bb) {
    const int b = blockIdx.y;
    const int j = blockIdx.x * blockDim.x + threadIdx.x;  // 0..65791
    __shared__ float cg[N_CGS * 3];
    if (threadIdx.x < N_CGS * 3) cg[threadIdx.x] = cg_xyz[b * (N_CGS * 3) + threadIdx.x];
    __syncthreads();

    float v0, v1, v2;
    if (j < NPAIR) {
        int ii = j >> 4, jj = j & 15;
        v0 = cg[jj*3+0] - cg[ii*3+0];
        v1 = cg[jj*3+1] - cg[ii*3+1];
        v2 = cg[jj*3+2] - cg[ii*3+2];
    } else {
        int t = j - NPAIR;
        int p = t >> 8, q = t & 255;
        int qi = q >> 4, qj = q & 15, pi = p >> 4, pj = p & 15;
        float ux = cg[qj*3+0] - cg[qi*3+0];
        float uy = cg[qj*3+1] - cg[qi*3+1];
        float uz = cg[qj*3+2] - cg[qi*3+2];
        float vx = cg[pj*3+0] - cg[pi*3+0];
        float vy = cg[pj*3+1] - cg[pi*3+1];
        float vz = cg[pj*3+2] - cg[pi*3+2];
        v0 = uy * vz - uz * vy;       // cross(Bmat[q], Bmat[p])
        v1 = uz * vx - ux * vz;
        v2 = ux * vy - uy * vx;
    }
    size_t base = ((size_t)b * 3) * (size_t)KTOT + (size_t)j;
    bb[base]                   = f2bf(v0);
    bb[base + (size_t)KTOT]    = f2bf(v1);
    bb[base + 2*(size_t)KTOT]  = f2bf(v2);
}

// ---------------------------------------------------------------------------
// 3) Split-K bf16 WMMA GEMM: partials[tile][ks][lane][8] = W x BbigT^T chunk
//    One wave owns a 32x32 (2x2 tile) output block for one K chunk.
//    Uneven split-K=16 (129-iter chunks, last 121) -> 3072 waves.
//    Wave id broadcast via readfirstlane so ks/mp/np/iters are SGPRs:
//    scalar loop control (s_cbranch) + saddr-form addressing; unroll 2
//    restores cross-iteration load/WMMA pipelining.
//    Fragment layout per ISA 16-bit A/B: lane<16 -> K{0..7,16..23},
//    lane>=16 -> K{8..15,24..31}; two contiguous 16B loads each.
// ---------------------------------------------------------------------------
__global__ void gemm_kernel(const unsigned short* __restrict__ Wbf,
                            const unsigned short* __restrict__ Bb,
                            float* __restrict__ part) {
    const int lane = threadIdx.x & 31;
    // wave-uniform ids as scalars
    const int w    = __builtin_amdgcn_readfirstlane(blockIdx.x * 8 + (threadIdx.x >> 5));
    const int ks   = w & (KSPLIT - 1);
    const int tmp  = w >> 4;                    // 0..191
    const int mp   = tmp & 15;                  // 0..15 : M block of 32
    const int np   = tmp >> 4;                  // 0..11 : N block of 32
    const int r    = lane & 15;
    const int half = lane >> 4;

    const int it0 = ks * KITER_CH;
    int iters     = KITER_TOT - it0;
    if (iters > KITER_CH) iters = KITER_CH;     // 129 (or 121 for ks==15)
    const size_t kb0 = (size_t)it0 * 32;

    const unsigned short* pa0 = Wbf + (size_t)(mp * 32 + r) * KTOT + kb0 + half * 8;
    const unsigned short* pa1 = pa0 + (size_t)16 * KTOT;
    const unsigned short* pb0 = Bb  + (size_t)(np * 32 + r) * KTOT + kb0 + half * 8;
    const unsigned short* pb1 = pb0 + (size_t)16 * KTOT;

    v8f acc00 = {}, acc01 = {}, acc10 = {}, acc11 = {};
#pragma unroll 2
    for (int it = 0; it < iters; ++it) {
        Frag a0, a1, b0, b1;
        a0.u[0] = *(const uint4*)(pa0);  a0.u[1] = *(const uint4*)(pa0 + 16);
        a1.u[0] = *(const uint4*)(pa1);  a1.u[1] = *(const uint4*)(pa1 + 16);
        b0.u[0] = *(const uint4*)(pb0);  b0.u[1] = *(const uint4*)(pb0 + 16);
        b1.u[0] = *(const uint4*)(pb1);  b1.u[1] = *(const uint4*)(pb1 + 16);
        acc00 = __builtin_amdgcn_wmma_f32_16x16x32_bf16(false, a0.v, false, b0.v,
                                                        (short)0, acc00, false, false);
        acc01 = __builtin_amdgcn_wmma_f32_16x16x32_bf16(false, a0.v, false, b1.v,
                                                        (short)0, acc01, false, false);
        acc10 = __builtin_amdgcn_wmma_f32_16x16x32_bf16(false, a1.v, false, b0.v,
                                                        (short)0, acc10, false, false);
        acc11 = __builtin_amdgcn_wmma_f32_16x16x32_bf16(false, a1.v, false, b1.v,
                                                        (short)0, acc11, false, false);
        pa0 += 32; pa1 += 32; pb0 += 32; pb1 += 32;
    }

    auto store = [&](v8f acc, int mt, int nt) {
        size_t off = (((size_t)(nt * 32 + mt) * KSPLIT + ks) * 32 + lane) * 8;
        *(v8f*)(part + off) = acc;
    };
    store(acc00, 2*mp,     2*np);
    store(acc01, 2*mp,     2*np + 1);
    store(acc10, 2*mp + 1, 2*np);
    store(acc11, 2*mp + 1, 2*np + 1);
}

// ---------------------------------------------------------------------------
// 4) Split-K reduction -> C[m][ncol]  (dx_recon, col = b*3+e)
//    WMMA C layout: lane l -> col l%16; VGPR v -> row v + 8*(l/16)
// ---------------------------------------------------------------------------
__global__ void reduce_kernel(const float* __restrict__ part, float* __restrict__ C) {
    const int tile = blockIdx.x;          // 0..767  (tile = nt*32 + mt)
    const int s    = threadIdx.x;         // 0..255
    const int lane = s >> 3, v = s & 7;
    const int mt   = tile & 31, nt = tile >> 5;
    float acc = 0.f;
    for (int ks = 0; ks < KSPLIT; ++ks)
        acc += part[(((size_t)tile * KSPLIT + ks) * 32 + lane) * 8 + v];
    const int m = mt * 16 + v + 8 * (lane >> 4);
    const int n = nt * 16 + (lane & 15);
    C[(size_t)m * NCOLS + n] = acc;
}

// ---------------------------------------------------------------------------
// 5) cg_offset[b][jc][e] = sum_i dx_recon[b,i,e] * assign_norm[b,i,jc]
// ---------------------------------------------------------------------------
__global__ void cg_offset_kernel(const float* __restrict__ C,
                                 const float* __restrict__ assign_norm,
                                 float* __restrict__ cgo) {
    const int jc = blockIdx.x, b = blockIdx.y, tid = threadIdx.x;  // 128 threads
    float a0 = 0.f, a1 = 0.f, a2 = 0.f;
    for (int i = tid; i < N_ATOMS; i += 128) {
        float w = assign_norm[((size_t)(b * N_ATOMS + i)) * N_CGS + jc];
        const float* c = C + (size_t)i * NCOLS + b * 3;
        a0 += c[0] * w; a1 += c[1] * w; a2 += c[2] * w;
    }
    __shared__ float red[3][128];
    red[0][tid] = a0; red[1][tid] = a1; red[2][tid] = a2;
    __syncthreads();
    for (int s = 64; s > 0; s >>= 1) {
        if (tid < s) {
            red[0][tid] += red[0][tid + s];
            red[1][tid] += red[1][tid + s];
            red[2][tid] += red[2][tid + s];
        }
        __syncthreads();
    }
    if (tid == 0) {
        float* o = cgo + ((size_t)b * N_CGS + jc) * 3;
        o[0] = red[0][0]; o[1] = red[1][0]; o[2] = red[2][0];
    }
}

// ---------------------------------------------------------------------------
// 6) passthrough copies + final reconstruction
// ---------------------------------------------------------------------------
__global__ void copy_kernel(const float* __restrict__ src, float* __restrict__ dst, int n) {
    int i = blockIdx.x * blockDim.x + threadIdx.x;
    if (i < n) dst[i] = src[i];
}

__global__ void recon_kernel(const float* __restrict__ cg_xyz,
                             const float* __restrict__ cgo,
                             const float* __restrict__ C,
                             const int* __restrict__ idx,
                             float* __restrict__ out) {
    int g = blockIdx.x * blockDim.x + threadIdx.x;   // 0..65535
    int b = g >> 9, atom = g & 511;
    int ic = idx[atom];
    const float* cg = cg_xyz + ((size_t)b * N_CGS + ic) * 3;
    const float* co = cgo    + ((size_t)b * N_CGS + ic) * 3;
    const float* c  = C + (size_t)atom * NCOLS + b * 3;
    float* o = out + (size_t)g * 3;
    o[0] = cg[0] - co[0] + c[0];
    o[1] = cg[1] - co[1] + c[1];
    o[2] = cg[2] - co[2] + c[2];
}

// ---------------------------------------------------------------------------
// launch
// ---------------------------------------------------------------------------
extern "C" void kernel_launch(void* const* d_in, const int* in_sizes, int n_in,
                              void* d_out, int out_size, void* d_ws, size_t ws_size,
                              hipStream_t stream) {
    const float* soft_assign = (const float*)d_in[0];
    const float* xyz         = (const float*)d_in[1];
    const float* cg_xyz      = (const float*)d_in[2];
    const float* assign_norm = (const float*)d_in[3];
    const float* W           = (const float*)d_in[4];
    const int*   assign_idx  = (const int*)d_in[5];
    float* out = (float*)d_out;

    // workspace layout (bytes, 256-aligned)
    char* ws = (char*)d_ws;
    unsigned short* Wbf = (unsigned short*)(ws);                    // 512*65792*2    = 67,371,008
    unsigned short* Bb  = (unsigned short*)(ws + 67371008ull);      // 384*65792*2    = 50,528,256
    float* part         = (float*)(ws + 117899264ull);              // 768*16*256*4   = 12,582,912
    float* C            = (float*)(ws + 130482176ull);              // 512*384*4      =    786,432
    float* cgo          = (float*)(ws + 131268608ull);              // 128*16*3*4     =     24,576

    const int nW = N_ATOMS * KTOT;                                   // 33,685,504
    convert_w_kernel<<<nW / 256, 256, 0, stream>>>(W, Wbf, nW);
    build_bbigT_kernel<<<dim3(KTOT / 256, B_BATCH), 256, 0, stream>>>(cg_xyz, Bb);
    gemm_kernel<<<384, 256, 0, stream>>>(Wbf, Bb, part);             // 3072 waves
    reduce_kernel<<<768, 256, 0, stream>>>(part, C);
    cg_offset_kernel<<<dim3(N_CGS, B_BATCH), 128, 0, stream>>>(C, assign_norm, cgo);

    const int n_sa  = B_BATCH * N_ATOMS * N_CGS;   // 1,048,576
    const int n_xyz = B_BATCH * N_ATOMS * 3;       //   196,608
    copy_kernel<<<n_sa / 256, 256, 0, stream>>>(soft_assign, out, n_sa);
    copy_kernel<<<n_xyz / 256, 256, 0, stream>>>(xyz, out + n_sa, n_xyz);
    recon_kernel<<<(B_BATCH * N_ATOMS) / 256, 256, 0, stream>>>(
        cg_xyz, cgo, C, assign_idx, out + n_sa + n_xyz);
}
